// SlowButCorrectMultiheadAttention_62053687492661
// MI455X (gfx1250) — compile-verified, matrix operand order checked by reading, not run on hardware
//
#include <hip/hip_runtime.h>
#include <hip/hip_bf16.h>

// ---------------------------------------------------------------------------
// Problem constants (from reference): S=2048, B=2, E=1024, H=16, D=64
// ---------------------------------------------------------------------------
#define SS 2048
#define BB 2
#define EE 1024
#define HH 16
#define DD 64
#define NN (SS * BB)   // 4096 token rows

typedef __attribute__((ext_vector_type(16))) __bf16        v16bf;
typedef __attribute__((ext_vector_type(8)))  float         v8f;
typedef __attribute__((ext_vector_type(4)))  unsigned int  v4u;
typedef __attribute__((ext_vector_type(4)))  int           v4i;

#define AS1 __attribute__((address_space(1)))
#define AS3 __attribute__((address_space(3)))

// f32 -> bf16 round-to-nearest-even
__device__ __forceinline__ unsigned short f2bf(float f) {
  unsigned int u = __builtin_bit_cast(unsigned int, f);
  u += 0x7FFFu + ((u >> 16) & 1u);
  return (unsigned short)(u >> 16);
}
__device__ __forceinline__ unsigned int pack2(float a, float b) {
  return (unsigned int)f2bf(a) | ((unsigned int)f2bf(b) << 16);
}

// ---------------------------------------------------------------------------
// Async global->LDS 16-byte copy (GLOBAL_LOAD_ASYNC_TO_LDS_B128, ASYNCcnt).
// ---------------------------------------------------------------------------
__device__ __forceinline__ void async_cp16(const unsigned short* g, unsigned short* l) {
#if __has_builtin(__builtin_amdgcn_global_load_async_to_lds_b128)
  __builtin_amdgcn_global_load_async_to_lds_b128((AS1 v4i*)g, (AS3 v4i*)l, 0, 0);
#else
  *(uint4*)l = *(const uint4*)g;
#endif
}

__device__ __forceinline__ void wait_async0() {
#if __has_builtin(__builtin_amdgcn_s_wait_asynccnt)
  __builtin_amdgcn_s_wait_asynccnt(0);
#else
  asm volatile("s_wait_asynccnt 0x0" ::: "memory");
#endif
}

// ---------------------------------------------------------------------------
// WMMA operand loader: lane l (r=l&15, half=l>>4) holds K = {half*8..+7,
// 16+half*8..+7} of row r of a row-major [16][stride] bf16 tile -> two b128s.
// ---------------------------------------------------------------------------
union OpCast { v4u u[2]; v16bf v; };

__device__ __forceinline__ v16bf load_op(const unsigned short* base, int stride) {
  const int lane = threadIdx.x & 31;
  const unsigned short* p = base + (size_t)(lane & 15) * stride + ((lane >> 4) << 3);
  OpCast oc;
  oc.u[0] = *(const v4u*)(p);
  oc.u[1] = *(const v4u*)(p + 16);
  return oc.v;
}

__device__ __forceinline__ v8f wmma_bf16(v16bf a, v16bf b, v8f c) {
  return __builtin_amdgcn_wmma_f32_16x16x32_bf16(false, a, false, b, (short)0, c,
                                                 false, false);
}

// ---------------------------------------------------------------------------
// Kernel 0: fp32 -> bf16 conversion (x and the four weight matrices) so all
// later staging is a pure byte copy the async-LDS engine can do.
// ---------------------------------------------------------------------------
__global__ void cvt_bf16_kernel(const float* __restrict__ in,
                                unsigned short* __restrict__ out) {
  const size_t i = ((size_t)blockIdx.x * blockDim.x + threadIdx.x) * 8;
  float4 f0 = *(const float4*)(in + i);
  float4 f1 = *(const float4*)(in + i + 4);
  uint4 p;
  p.x = pack2(f0.x, f0.y); p.y = pack2(f0.z, f0.w);
  p.z = pack2(f1.x, f1.y); p.w = pack2(f1.z, f1.w);
  *(uint4*)(out + i) = p;
}

// ---------------------------------------------------------------------------
// Shared GEMM mainloop: C[128x64] tile, block = 256 threads (8 waves).
// Wave w owns rows w*16..w*16+15 and 4 column tiles (4 accumulators = 32
// VGPRs -> no spills).  LDS double-buffered; per K-step: 3 async b128 issues
// (per-lane pointers hoisted and bumped by +32), 1 A-op + 4 B-ops, 4 WMMAs.
// ---------------------------------------------------------------------------
#define A_TILE (128 * 32)
#define B_TILE (64 * 32)

__device__ __forceinline__ void gemm_mainloop(const unsigned short* gA,   // += n0 rows
                                              const unsigned short* gB,   // += m0 rows
                                              unsigned short* As, unsigned short* Bs,
                                              v8f c[4]) {
  const int tid = threadIdx.x;
  const int wid = tid >> 5;
  const int oa0 = tid * 8;            // A shot 0: rows 0..63
  const int oa1 = (256 + tid) * 8;    // A shot 1: rows 64..127
  const int ob0 = tid * 8;            // B shot: rows 0..63

  const unsigned short* pA0 = gA + (size_t)(oa0 >> 5) * EE + (oa0 & 31);
  const unsigned short* pA1 = gA + (size_t)(oa1 >> 5) * EE + (oa1 & 31);
  const unsigned short* pB0 = gB + (size_t)(ob0 >> 5) * EE + (ob0 & 31);

  async_cp16(pA0, As + oa0);
  async_cp16(pA1, As + oa1);
  async_cp16(pB0, Bs + ob0);

  for (int it = 0; it < EE / 32; ++it) {
    wait_async0();
    __syncthreads();
    const int cur = it & 1, nxt = cur ^ 1;
    if (it + 1 < EE / 32) {
      pA0 += 32; pA1 += 32; pB0 += 32;
      async_cp16(pA0, As + nxt * A_TILE + oa0);
      async_cp16(pA1, As + nxt * A_TILE + oa1);
      async_cp16(pB0, Bs + nxt * B_TILE + ob0);
    }
    const unsigned short* Ac = As + cur * A_TILE;
    const unsigned short* Bc = Bs + cur * B_TILE;
    v16bf a = load_op(Ac + wid * 16 * 32, 32);
    for (int ct = 0; ct < 4; ++ct) {
      v16bf b = load_op(Bc + ct * 16 * 32, 32);
      c[ct] = wmma_bf16(a, b, c[ct]);
    }
  }
}

// ---------------------------------------------------------------------------
// Kernel 1: QKV projection.  C[n,m] = sum_k xb[n,k]*Wb[m,k] + bias[m].
// Outputs: q,k bf16 [b][h][s][d]; vT bf16 [b][h][d][s].
// ---------------------------------------------------------------------------
__global__ void qkv_proj_kernel(const unsigned short* __restrict__ xb,
                                const unsigned short* __restrict__ wqb,
                                const unsigned short* __restrict__ wkb,
                                const unsigned short* __restrict__ wvb,
                                const float* __restrict__ bq, const float* __restrict__ bk,
                                const float* __restrict__ bv,
                                unsigned short* __restrict__ q, unsigned short* __restrict__ k,
                                unsigned short* __restrict__ vt) {
  const int which = blockIdx.z;                         // uniform
  const unsigned short* W = (which == 0) ? wqb : (which == 1) ? wkb : wvb;
  const float* bias       = (which == 0) ? bq  : (which == 1) ? bk  : bv;

  const int m0 = blockIdx.x * 64;
  const int n0 = blockIdx.y * 128;
  const int lane = threadIdx.x & 31, wid = threadIdx.x >> 5;

  __shared__ __align__(16) unsigned short As[2 * A_TILE];
  __shared__ __align__(16) unsigned short Bs[2 * B_TILE];

  v8f c[4] = {{}, {}, {}, {}};
  gemm_mainloop(xb + (size_t)n0 * EE, W + (size_t)m0 * EE, As, Bs, c);

  const int half = lane >> 4, nloc = lane & 15;
  for (int ct = 0; ct < 4; ++ct) {
    const int m = m0 + ct * 16 + nloc;
    const float bval = bias[m];
    const int hh = m >> 6, dd = m & 63;
    for (int j = 0; j < 8; ++j) {
      const int n = n0 + wid * 16 + j + 8 * half;      // flat token row = s*B + b
      const int s = n >> 1, bidx = n & 1;
      const unsigned short obf = f2bf(c[ct][j] + bval);
      if (which == 2) {
        vt[(((size_t)(bidx * HH + hh) * DD) + dd) * SS + s] = obf;
      } else {
        unsigned short* dst = (which == 0) ? q : k;
        dst[(((size_t)(bidx * HH + hh) * SS) + s) * DD + dd] = obf;
      }
    }
  }
}

// ---------------------------------------------------------------------------
// Kernel 2 (pass A): per-(s,b,t) max & sum-of-exp over the 16 heads.
// Head-axis softmax statistics are purely elementwise across the 16 WMMA
// accumulators (identical (VGPR,lane) positions) -> no cross-lane traffic.
// ---------------------------------------------------------------------------
__global__ void attn_ml_kernel(const unsigned short* __restrict__ qb,
                               const unsigned short* __restrict__ kb,
                               float* __restrict__ mb, float* __restrict__ lb) {
  const int s0 = blockIdx.x * 16;
  const int b  = blockIdx.y;
  const int tid = threadIdx.x, lane = tid & 31, wid = tid >> 5;

  __shared__ __align__(16) unsigned short qs[16 * EE];  // qs[r][h*64+d]

  for (int s = 0; s < 8; ++s) {           // 16384 ushorts = 8 shots x 256 x 16B
    const int ci = s * 256 + tid;
    const int h = ci >> 7, rem = ci & 127, r = rem >> 3, dc = (rem & 7) * 8;
    async_cp16(qb + (((size_t)(b * HH + h) * SS + s0 + r) * DD + dc),
               qs + r * EE + h * DD + dc);
  }
  wait_async0();
  __syncthreads();

  const int half = lane >> 4, nloc = lane & 15;

  for (int ti = wid; ti < SS / 16; ti += 8) {
    const int t0 = ti * 16;
    v8f sc[HH];
    for (int h = 0; h < HH; ++h) {
      v16bf a0 = load_op(qs + h * DD, EE);
      v16bf a1 = load_op(qs + h * DD + 32, EE);
      const unsigned short* kr = kb + ((size_t)(b * HH + h) * SS + t0) * DD;
      v16bf b0 = load_op(kr, DD);
      v16bf b1 = load_op(kr + 32, DD);
      v8f c = {};
      c = wmma_bf16(a0, b0, c);
      c = wmma_bf16(a1, b1, c);
      sc[h] = c;
    }
    for (int j = 0; j < 8; ++j) {
      float mx = sc[0][j] * 0.125f;
      for (int h = 1; h < HH; ++h) mx = fmaxf(mx, sc[h][j] * 0.125f);
      float sum = 0.f;
      for (int h = 0; h < HH; ++h) sum += __expf(sc[h][j] * 0.125f - mx);
      const int srow = s0 + j + 8 * half;
      const size_t idx = ((size_t)b * SS + srow) * SS + t0 + nloc;
      mb[idx] = mx;
      lb[idx] = sum;
    }
  }
}

// ---------------------------------------------------------------------------
// Kernel 3 (pass B): per-(b,h) out = normalized(scores) @ v; score tiles are
// recomputed, normalized with (m,l), converted C-layout->A-layout via a 1 KB
// per-wave LDS round trip, and accumulated over K(t)=2048 with WMMA.
// ---------------------------------------------------------------------------
__global__ void attn_av_kernel(const unsigned short* __restrict__ qb,
                               const unsigned short* __restrict__ kb,
                               const unsigned short* __restrict__ vtb,
                               const float* __restrict__ mb,
                               const float* __restrict__ lb,
                               unsigned short* __restrict__ ab) {
  const int b = blockIdx.z, h = blockIdx.y;
  const int tid = threadIdx.x, lane = tid & 31, wid = tid >> 5;
  const int s0 = (blockIdx.x * 8 + wid) * 16;

  __shared__ __align__(16) unsigned short att[8][16 * 32];
  unsigned short* at = att[wid];

  const size_t bh = (size_t)(b * HH + h);
  const unsigned short* qr = qb + (bh * SS + s0) * DD;
  const v16bf aq0 = load_op(qr, DD);
  const v16bf aq1 = load_op(qr + 32, DD);

  const int half = lane >> 4, nloc = lane & 15;
  v8f o[4] = {{}, {}, {}, {}};

  for (int tt = 0; tt < SS / 32; ++tt) {
    const int t0 = tt * 32;
    if (tt + 1 < SS / 32) {   // prefetch next k / v stripes into caches
      __builtin_prefetch(kb + (bh * SS + t0 + 32) * DD, 0, 3);
      __builtin_prefetch(vtb + bh * DD * SS + t0 + 32, 0, 3);
    }
    for (int sub = 0; sub < 2; ++sub) {
      const int t16 = t0 + sub * 16;
      const unsigned short* kr = kb + (bh * SS + t16) * DD;
      v16bf b0 = load_op(kr, DD);
      v16bf b1 = load_op(kr + 32, DD);
      v8f sc = {};
      sc = wmma_bf16(aq0, b0, sc);
      sc = wmma_bf16(aq1, b1, sc);
      const int t = t16 + nloc;
      for (int j = 0; j < 8; ++j) {
        const int srow = s0 + j + 8 * half;
        const size_t idx = ((size_t)b * SS + srow) * SS + t;
        const float p = __expf(sc[j] * 0.125f - mb[idx]) / lb[idx];
        at[(j + 8 * half) * 32 + sub * 16 + nloc] = f2bf(p);
      }
    }
    asm volatile("s_wait_dscnt 0x0" ::: "memory");  // wave-local LDS RAW
    v16bf aa = load_op(at, 32);
    for (int dt = 0; dt < 4; ++dt) {
      v16bf bv = load_op(vtb + (bh * DD + dt * 16) * SS + t0, SS);
      o[dt] = wmma_bf16(aa, bv, o[dt]);
    }
    asm volatile("s_wait_dscnt 0x0" ::: "memory");
  }

  for (int dt = 0; dt < 4; ++dt)
    for (int j = 0; j < 8; ++j) {
      const int srow = s0 + j + 8 * half;
      const int d = dt * 16 + nloc;
      ab[((size_t)srow * BB + b) * EE + h * DD + d] = f2bf(o[dt][j]);
    }
}

// ---------------------------------------------------------------------------
// Kernel 4: output projection, fp32 result to d_out.
// ---------------------------------------------------------------------------
__global__ void out_proj_kernel(const unsigned short* __restrict__ A,
                                const unsigned short* __restrict__ wob,
                                const float* __restrict__ bo,
                                float* __restrict__ out) {
  const int m0 = blockIdx.x * 64;
  const int n0 = blockIdx.y * 128;
  const int lane = threadIdx.x & 31, wid = threadIdx.x >> 5;

  __shared__ __align__(16) unsigned short As[2 * A_TILE];
  __shared__ __align__(16) unsigned short Bs[2 * B_TILE];

  v8f c[4] = {{}, {}, {}, {}};
  gemm_mainloop(A + (size_t)n0 * EE, wob + (size_t)m0 * EE, As, Bs, c);

  const int half = lane >> 4, nloc = lane & 15;
  for (int ct = 0; ct < 4; ++ct) {
    const int m = m0 + ct * 16 + nloc;
    const float bval = bo[m];
    for (int j = 0; j < 8; ++j) {
      const int n = n0 + wid * 16 + j + 8 * half;
      out[(size_t)n * EE + m] = c[ct][j] + bval;
    }
  }
}

// ---------------------------------------------------------------------------
// Launch
// ---------------------------------------------------------------------------
extern "C" void kernel_launch(void* const* d_in, const int* in_sizes, int n_in,
                              void* d_out, int out_size, void* d_ws, size_t ws_size,
                              hipStream_t stream) {
  const float* x  = (const float*)d_in[0];
  const float* Wq = (const float*)d_in[1];
  const float* bq = (const float*)d_in[2];
  const float* Wk = (const float*)d_in[3];
  const float* bk = (const float*)d_in[4];
  const float* Wv = (const float*)d_in[5];
  const float* bv = (const float*)d_in[6];
  const float* Wo = (const float*)d_in[7];
  const float* bo = (const float*)d_in[8];
  float* out = (float*)d_out;

  // Workspace layout (bytes)
  const size_t XB  = (size_t)NN * EE * 2;                 // 8 MB  bf16 x
  const size_t WB  = (size_t)EE * EE * 2;                 // 2 MB  bf16 weight
  const size_t QKB = (size_t)BB * HH * SS * DD * 2;       // 8 MB  q / k / vT
  const size_t MLB = (size_t)BB * SS * SS * 4;            // 33.5 MB m / l

  char* ws = (char*)d_ws;
  unsigned short* xb   = (unsigned short*)(ws);
  unsigned short* wqb  = (unsigned short*)(ws + XB);
  unsigned short* wkb  = (unsigned short*)(ws + XB + WB);
  unsigned short* wvb  = (unsigned short*)(ws + XB + 2 * WB);
  unsigned short* wob  = (unsigned short*)(ws + XB + 3 * WB);
  unsigned short* qbuf = (unsigned short*)(ws + XB + 4 * WB);
  unsigned short* kbuf = (unsigned short*)(ws + XB + 4 * WB + QKB);
  unsigned short* vtb  = (unsigned short*)(ws + XB + 4 * WB + 2 * QKB);
  float*          mbuf = (float*)(ws + XB + 4 * WB + 3 * QKB);
  float*          lbuf = (float*)(ws + XB + 4 * WB + 3 * QKB + MLB);
  unsigned short* aout = (unsigned short*)(ws + XB + 4 * WB + 3 * QKB + 2 * MLB);

  // 0) one-shot fp32 -> bf16 copies (makes all staging async-copyable)
  cvt_bf16_kernel<<<(NN * EE) / (256 * 8), 256, 0, stream>>>(x, xb);
  cvt_bf16_kernel<<<(EE * EE) / (256 * 8), 256, 0, stream>>>(Wq, wqb);
  cvt_bf16_kernel<<<(EE * EE) / (256 * 8), 256, 0, stream>>>(Wk, wkb);
  cvt_bf16_kernel<<<(EE * EE) / (256 * 8), 256, 0, stream>>>(Wv, wvb);
  cvt_bf16_kernel<<<(EE * EE) / (256 * 8), 256, 0, stream>>>(Wo, wob);

  // 1) QKV projections (bf16 WMMA, fp32 accum, async-LDS double buffering)
  qkv_proj_kernel<<<dim3(EE / 64, NN / 128, 3), 256, 0, stream>>>(
      xb, wqb, wkb, wvb, bq, bk, bv, qbuf, kbuf, vtb);

  // 2) head-axis logsumexp statistics
  attn_ml_kernel<<<dim3(SS / 16, BB), 256, 0, stream>>>(qbuf, kbuf, mbuf, lbuf);

  // 3) normalized attention @ v
  attn_av_kernel<<<dim3(SS / 128, HH, BB), 256, 0, stream>>>(
      qbuf, kbuf, vtb, mbuf, lbuf, aout);

  // 4) output projection
  out_proj_kernel<<<dim3(EE / 64, NN / 128), 256, 0, stream>>>(aout, wob, bo, out);
}
